// HierarchicalTemporalAttention_42408507080888
// MI455X (gfx1250) — matrix operand order repaired and checked by reading, not compiled, not run on hardware
//
#include <hip/hip_runtime.h>
#include <hip/hip_fp16.h>

typedef _Float16 h16;
typedef __attribute__((ext_vector_type(16))) _Float16 v16h;
typedef __attribute__((ext_vector_type(8)))  _Float16 v8h;
typedef __attribute__((ext_vector_type(8)))  float    v8f;

// ---------------------------------------------------------------------------
// WMMA fragment helpers (gfx1250 wave32 layouts, CDNA5 ISA 7.12.2)
// A 16x32 f16: lane L (M=L&15, g=L>>4): elems 0..7 -> K=8g+0..7, 8..15 -> K=8g+16..23
// B 32x16 f16 from Bt[N,K] row-major: lane L (N=L&15, g=L>>4): elems 0..7 -> K=8g+0..7, 8..15 -> K=8g+16..23
// C/D 16x16 f32: lane L elem r -> (M = r + 8*(L>>4), N = L&15)
// ---------------------------------------------------------------------------
__device__ __forceinline__ v16h frag_cat(v8h lo, v8h hi) {
  return __builtin_shufflevector(lo, hi, 0,1,2,3,4,5,6,7,8,9,10,11,12,13,14,15);
}

__device__ __forceinline__ v16h load_a16(const h16* A, int lda, int mBase, int k) {
  int lane = threadIdx.x & 31;
  const h16* p = A + (size_t)(mBase + (lane & 15)) * lda + k + 8 * (lane >> 4);
  v8h lo = *(const v8h*)p;
  v8h hi = *(const v8h*)(p + 16);
  return frag_cat(lo, hi);
}

__device__ __forceinline__ v16h load_b16(const h16* Bt, int ldb, int nBase, int k) {
  int lane = threadIdx.x & 31;
  const h16* p = Bt + (size_t)(nBase + (lane & 15)) * ldb + k + 8 * (lane >> 4);
  v8h lo = *(const v8h*)p;
  v8h hi = *(const v8h*)(p + 16);
  return frag_cat(lo, hi);
}

__device__ __forceinline__ v8f wmma32(v16h a, v16h b, v8f c) {
  return __builtin_amdgcn_wmma_f32_16x16x32_f16(false, a, false, b, (short)0, c,
                                                false, false);
}

// ---------------------------------------------------------------------------
// Generic f16 GEMM: C[M,N] = A[M,K] @ Bt[N,K]^T + bias[N]
// Block 256 thr = 8 waves, block tile 128x128, wave tile 32x64.
// Shapes used here are all multiples of 128/32 -> no edge handling.
// ---------------------------------------------------------------------------
__global__ __launch_bounds__(256) void gemm_f16_kernel(
    const h16* __restrict__ A, int lda,
    const h16* __restrict__ Bt, int ldb,
    const float* __restrict__ bias,
    void* __restrict__ Cout, int ldc, int coff, int outF32, int K)
{
  int wave = threadIdx.x >> 5;
  int lane = threadIdx.x & 31;
  int m0 = blockIdx.y * 128 + (wave & 3) * 32;
  int n0 = blockIdx.x * 128 + (wave >> 2) * 64;
  v8f zero = {};
  v8f acc[2][4];
#pragma unroll
  for (int i = 0; i < 2; ++i)
#pragma unroll
    for (int j = 0; j < 4; ++j) acc[i][j] = zero;

  for (int k = 0; k < K; k += 32) {
    v16h a0 = load_a16(A, lda, m0, k);
    v16h a1 = load_a16(A, lda, m0 + 16, k);
#pragma unroll
    for (int j = 0; j < 4; ++j) {
      v16h b = load_b16(Bt, ldb, n0 + 16 * j, k);
      acc[0][j] = wmma32(a0, b, acc[0][j]);
      acc[1][j] = wmma32(a1, b, acc[1][j]);
    }
  }
  int colL = lane & 15;
  int rowOfs = 8 * (lane >> 4);
#pragma unroll
  for (int i = 0; i < 2; ++i)
#pragma unroll
    for (int j = 0; j < 4; ++j) {
      int col = n0 + 16 * j + colL;
      float bv = bias ? bias[col] : 0.f;
#pragma unroll
      for (int r = 0; r < 8; ++r) {
        int row = m0 + 16 * i + rowOfs + r;
        float v = acc[i][j][r] + bv;
        size_t idx = (size_t)row * ldc + coff + col;
        if (outF32) ((float*)Cout)[idx] = v;
        else        ((h16*)Cout)[idx]   = (h16)v;
      }
    }
}

// ---------------------------------------------------------------------------
// Local windowed attention (w in {3,5,7}) with the "transpose-free" head split.
// One block per token; wave h handles weird-head h; lane owns 2 of the 64 d2.
// Padded rows of x project to the bias vector. Output = mean over window rows,
// recomposed deterministically through LDS (no atomics).
// ---------------------------------------------------------------------------
template <int W>
__global__ __launch_bounds__(256) void local_attn_kernel(
    const h16* __restrict__ Qp, const h16* __restrict__ Kp,
    const h16* __restrict__ Vp,
    const float* __restrict__ bq, const float* __restrict__ bk,
    const float* __restrict__ bv,
    h16* __restrict__ meanatt)
{
  const int PAD = (W - 1) / 2;
  __shared__ float oall[512 * W];
  int tid = threadIdx.x;
  int token = blockIdx.x;
  int b = token >> 10, t = token & 1023;
  int h = tid >> 5, lane = tid & 31;
  int c0 = lane * 2;

  float q[W][2], kk[W][2], vv[W][2];
#pragma unroll
  for (int t2 = 0; t2 < W; ++t2) {
#pragma unroll
    for (int u = 0; u < 2; ++u) {
      int f = h * W * 64 + t2 * 64 + c0 + u;
      int j = f >> 9, ch = f & 511;
      int s = t - PAD + j;
      bool in = (s >= 0) && (s < 1024);
      size_t off = ((size_t)(b * 1024 + (in ? s : 0))) * 512 + ch;
      q[t2][u]  = in ? (float)Qp[off] : bq[ch];
      kk[t2][u] = in ? (float)Kp[off] : bk[ch];
      vv[t2][u] = in ? (float)Vp[off] : bv[ch];
    }
  }

  float o[W][2];
#pragma unroll
  for (int i = 0; i < W; ++i) { o[i][0] = 0.f; o[i][1] = 0.f; }

#pragma unroll
  for (int i = 0; i < W; ++i) {
    float sc[W];
#pragma unroll
    for (int j = 0; j < W; ++j) {
      float p = q[i][0] * kk[j][0] + q[i][1] * kk[j][1];
#pragma unroll
      for (int m = 16; m >= 1; m >>= 1) p += __shfl_xor(p, m, 32);
      sc[j] = p * 0.125f;  // 1/sqrt(64)
    }
    float mx = sc[0];
#pragma unroll
    for (int j = 1; j < W; ++j) mx = fmaxf(mx, sc[j]);
    float sum = 0.f;
#pragma unroll
    for (int j = 0; j < W; ++j) { sc[j] = __expf(sc[j] - mx); sum += sc[j]; }
    float inv = 1.f / sum;
#pragma unroll
    for (int j = 0; j < W; ++j) {
      float wt = sc[j] * inv;
      o[i][0] += wt * vv[j][0];
      o[i][1] += wt * vv[j][1];
    }
  }

#pragma unroll
  for (int t2 = 0; t2 < W; ++t2) {
    oall[h * W * 64 + t2 * 64 + c0]     = o[t2][0];
    oall[h * W * 64 + t2 * 64 + c0 + 1] = o[t2][1];
  }
  __syncthreads();
  const float invw = 1.f / (float)W;
  for (int d = tid; d < 512; d += 256) {
    float s = 0.f;
#pragma unroll
    for (int r = 0; r < W; ++r) s += oall[d + 512 * r];
    meanatt[(size_t)token * 512 + d] = (h16)(s * invw);
  }
}

// ---------------------------------------------------------------------------
// Global attention, flash style. The buggy reshape makes each of the 64 groups
// (b,h) a contiguous [1024,64] f16 slice. One wave handles 16 q-rows; keys
// processed 32 at a time; P restaged via per-wave LDS (DS ops in-order/wave).
// ---------------------------------------------------------------------------
__global__ __launch_bounds__(256) void flash_global_kernel(
    const h16* __restrict__ Qg, const h16* __restrict__ Kg,
    const h16* __restrict__ Vt, h16* __restrict__ attg)
{
  __shared__ __align__(16) h16 Pst[8][512];  // per-wave 16x32 staging
  int wave = threadIdx.x >> 5;
  int lane = threadIdx.x & 31;
  int grp = blockIdx.x >> 3;
  int qBase = (blockIdx.x & 7) * 128 + wave * 16;
  const h16* Q   = Qg + (size_t)grp * 65536;
  const h16* Km  = Kg + (size_t)grp * 65536;
  const h16* Vtg = Vt + (size_t)grp * 65536;  // [64][1024]
  h16* Pw = &Pst[wave][0];
  int colL = lane & 15;
  int half = lane >> 4;

  v8f zero = {};
  v8f o0 = zero, o1 = zero, o2 = zero, o3 = zero;
  float m[8], l[8];
#pragma unroll
  for (int r = 0; r < 8; ++r) { m[r] = -1e30f; l[r] = 0.f; }

  v16h aq0 = load_a16(Q, 64, qBase, 0);
  v16h aq1 = load_a16(Q, 64, qBase, 32);

  for (int kb = 0; kb < 1024; kb += 32) {
    v8f s0 = zero, s1 = zero;
    s0 = wmma32(aq0, load_b16(Km, 64, kb, 0),       s0);
    s0 = wmma32(aq1, load_b16(Km, 64, kb, 32),      s0);
    s1 = wmma32(aq0, load_b16(Km, 64, kb + 16, 0),  s1);
    s1 = wmma32(aq1, load_b16(Km, 64, kb + 16, 32), s1);

    float p0[8], p1[8];
#pragma unroll
    for (int r = 0; r < 8; ++r) { p0[r] = s0[r] * 0.125f; p1[r] = s1[r] * 0.125f; }
#pragma unroll
    for (int r = 0; r < 8; ++r) {
      float bm = fmaxf(p0[r], p1[r]);
#pragma unroll
      for (int msk = 8; msk >= 1; msk >>= 1) bm = fmaxf(bm, __shfl_xor(bm, msk, 16));
      float mn = fmaxf(m[r], bm);
      float sc = __expf(m[r] - mn);
      m[r] = mn;
      l[r] *= sc;
      o0[r] *= sc; o1[r] *= sc; o2[r] *= sc; o3[r] *= sc;
      p0[r] = __expf(p0[r] - mn);
      p1[r] = __expf(p1[r] - mn);
      float rs = p0[r] + p1[r];
#pragma unroll
      for (int msk = 8; msk >= 1; msk >>= 1) rs += __shfl_xor(rs, msk, 16);
      l[r] += rs;
    }
    // stage P (16 q x 32 k) row-major in per-wave LDS, reload as A fragment
#pragma unroll
    for (int r = 0; r < 8; ++r) {
      int row = r + 8 * half;
      Pw[row * 32 + colL]      = (h16)p0[r];
      Pw[row * 32 + 16 + colL] = (h16)p1[r];
    }
    v16h pa = load_a16(Pw, 32, 0, 0);
    o0 = wmma32(pa, load_b16(Vtg, 1024, 0,  kb), o0);
    o1 = wmma32(pa, load_b16(Vtg, 1024, 16, kb), o1);
    o2 = wmma32(pa, load_b16(Vtg, 1024, 32, kb), o2);
    o3 = wmma32(pa, load_b16(Vtg, 1024, 48, kb), o3);
  }
#pragma unroll
  for (int r = 0; r < 8; ++r) {
    int srow = qBase + r + 8 * half;
    float invl = 1.f / l[r];
    size_t base = (size_t)grp * 65536 + (size_t)srow * 64;
    attg[base +  0 + colL] = (h16)(o0[r] * invl);
    attg[base + 16 + colL] = (h16)(o1[r] * invl);
    attg[base + 32 + colL] = (h16)(o2[r] * invl);
    attg[base + 48 + colL] = (h16)(o3[r] * invl);
  }
}

// ---------------------------------------------------------------------------
// Small utility kernels
// ---------------------------------------------------------------------------
__global__ void f32_to_f16_kernel(const float* __restrict__ in,
                                  h16* __restrict__ out, int n) {
  int i = blockIdx.x * 256 + threadIdx.x;
  if (i < n) out[i] = (h16)in[i];
}

// Wt[n*K + k] = (f16) W[k*N + n]
__global__ void transpose_w_kernel(const float* __restrict__ W,
                                   h16* __restrict__ Wt, int K, int N) {
  int idx = blockIdx.x * 256 + threadIdx.x;
  if (idx >= K * N) return;
  int n = idx / K, k = idx % K;
  Wt[idx] = (h16)W[(size_t)k * N + n];
}

// Per-group [1024,64] -> [64,1024] f16 transpose (64 groups)
__global__ void transpose_v_kernel(const h16* __restrict__ Vg,
                                   h16* __restrict__ Vt) {
  int g = blockIdx.x;
  const h16* src = Vg + (size_t)g * 65536;
  h16* dst = Vt + (size_t)g * 65536;
  for (int idx = threadIdx.x; idx < 65536; idx += blockDim.x) {
    int s = idx >> 6, d = idx & 63;
    dst[d * 1024 + s] = src[idx];
  }
}

// ---------------------------------------------------------------------------
extern "C" void kernel_launch(void* const* d_in, const int* in_sizes, int n_in,
                              void* d_out, int out_size, void* d_ws,
                              size_t ws_size, hipStream_t stream) {
  (void)in_sizes; (void)n_in; (void)out_size; (void)ws_size;
  const float* x      = (const float*)d_in[0];
  const float* loc_wq = (const float*)d_in[1];
  const float* loc_bq = (const float*)d_in[2];
  const float* loc_wk = (const float*)d_in[3];
  const float* loc_bk = (const float*)d_in[4];
  const float* loc_wv = (const float*)d_in[5];
  const float* loc_bv = (const float*)d_in[6];
  const float* loc_wo = (const float*)d_in[7];
  const float* loc_bo = (const float*)d_in[8];
  const float* g_wq   = (const float*)d_in[9];
  const float* g_bq   = (const float*)d_in[10];
  const float* g_wk   = (const float*)d_in[11];
  const float* g_bk   = (const float*)d_in[12];
  const float* g_wv   = (const float*)d_in[13];
  const float* g_bv   = (const float*)d_in[14];
  const float* g_wo   = (const float*)d_in[15];
  const float* g_bo   = (const float*)d_in[16];
  const float* out_w  = (const float*)d_in[17];
  const float* out_b  = (const float*)d_in[18];

  const int M = 8192;           // B*T
  const size_t XE = (size_t)M * 512;

  char* ws = (char*)d_ws;
  size_t off = 0;
  auto alloc = [&](size_t bytes) -> void* {
    void* p = ws + off;
    off += (bytes + 255) & ~(size_t)255;
    return p;
  };
  h16* x16 = (h16*)alloc(XE * 2);
  h16* wq_t[3], *wk_t[3], *wv_t[3], *wo_t[3];
  for (int lv = 0; lv < 3; ++lv) {
    wq_t[lv] = (h16*)alloc(262144 * 2);
    wk_t[lv] = (h16*)alloc(262144 * 2);
    wv_t[lv] = (h16*)alloc(262144 * 2);
    wo_t[lv] = (h16*)alloc(262144 * 2);
  }
  h16* gq_t = (h16*)alloc(262144 * 2);
  h16* gk_t = (h16*)alloc(262144 * 2);
  h16* gv_t = (h16*)alloc(262144 * 2);
  h16* go_t = (h16*)alloc(262144 * 2);
  h16* ow_t = (h16*)alloc((size_t)2048 * 512 * 2);
  h16* Qp  = (h16*)alloc(XE * 2);
  h16* Kp  = (h16*)alloc(XE * 2);
  h16* Vp  = (h16*)alloc(XE * 2);
  h16* Vt  = (h16*)alloc(XE * 2);
  h16* matt = (h16*)alloc(XE * 2);
  h16* attg = (h16*)alloc(XE * 2);
  h16* comb = (h16*)alloc((size_t)M * 2048 * 2);

  auto gemm = [&](const h16* A, int lda, const h16* Bt, int ldb,
                  const float* bias, void* C, int ldc, int coff, int outF32,
                  int Mm, int Nn, int Kk) {
    dim3 grid(Nn / 128, Mm / 128);
    gemm_f16_kernel<<<grid, 256, 0, stream>>>(A, lda, Bt, ldb, bias, C, ldc,
                                              coff, outF32, Kk);
  };
  auto transW = [&](const float* W, h16* Wt_, int Kk, int Nn) {
    int total = Kk * Nn;
    transpose_w_kernel<<<(total + 255) / 256, 256, 0, stream>>>(W, Wt_, Kk, Nn);
  };

  // 0) convert x, transpose+convert all weights
  f32_to_f16_kernel<<<(int)((XE + 255) / 256), 256, 0, stream>>>(x, x16, (int)XE);
  for (int lv = 0; lv < 3; ++lv) {
    transW(loc_wq + (size_t)lv * 262144, wq_t[lv], 512, 512);
    transW(loc_wk + (size_t)lv * 262144, wk_t[lv], 512, 512);
    transW(loc_wv + (size_t)lv * 262144, wv_t[lv], 512, 512);
    transW(loc_wo + (size_t)lv * 262144, wo_t[lv], 512, 512);
  }
  transW(g_wq, gq_t, 512, 512);
  transW(g_wk, gk_t, 512, 512);
  transW(g_wv, gv_t, 512, 512);
  transW(g_wo, go_t, 512, 512);
  transW(out_w, ow_t, 2048, 512);

  // 1) local levels
  for (int lv = 0; lv < 3; ++lv) {
    gemm(x16, 512, wq_t[lv], 512, loc_bq + lv * 512, Qp, 512, 0, 0, M, 512, 512);
    gemm(x16, 512, wk_t[lv], 512, loc_bk + lv * 512, Kp, 512, 0, 0, M, 512, 512);
    gemm(x16, 512, wv_t[lv], 512, loc_bv + lv * 512, Vp, 512, 0, 0, M, 512, 512);
    if (lv == 0)
      local_attn_kernel<3><<<M, 256, 0, stream>>>(Qp, Kp, Vp, loc_bq, loc_bk,
                                                  loc_bv, matt);
    else if (lv == 1)
      local_attn_kernel<5><<<M, 256, 0, stream>>>(Qp, Kp, Vp, loc_bq + 512,
                                                  loc_bk + 512, loc_bv + 512, matt);
    else
      local_attn_kernel<7><<<M, 256, 0, stream>>>(Qp, Kp, Vp, loc_bq + 1024,
                                                  loc_bk + 1024, loc_bv + 1024, matt);
    gemm(matt, 512, wo_t[lv], 512, loc_bo + lv * 512, comb, 2048, lv * 512, 0,
         M, 512, 512);
  }

  // 2) global attention
  gemm(x16, 512, gq_t, 512, g_bq, Qp, 512, 0, 0, M, 512, 512);
  gemm(x16, 512, gk_t, 512, g_bk, Kp, 512, 0, 0, M, 512, 512);
  gemm(x16, 512, gv_t, 512, g_bv, Vp, 512, 0, 0, M, 512, 512);
  transpose_v_kernel<<<64, 256, 0, stream>>>(Vp, Vt);
  flash_global_kernel<<<512, 256, 0, stream>>>(Qp, Kp, Vt, attg);
  gemm(attg, 512, go_t, 512, g_bo, comb, 2048, 3 * 512, 0, M, 512, 512);

  // 3) final projection: [8192,2048] @ out_w + out_b -> f32 d_out
  gemm(comb, 2048, ow_t, 2048, out_b, d_out, 512, 0, 1, M, 512, 2048);
}
